// CEGAN_87703232184761
// MI455X (gfx1250) — matrix operand order repaired and checked by reading, not compiled, not run on hardware
//
#include <hip/hip_runtime.h>

// ---------------------------------------------------------------------------
// CEGAN forward for MI455X (gfx1250): f16 WMMA 16x16x32 for all GEMMs,
// one node per 256-thread block (8 wave32), LDS staged via async-to-LDS copies.
// ---------------------------------------------------------------------------

typedef __attribute__((ext_vector_type(16))) _Float16 v16h;
typedef __attribute__((ext_vector_type(8)))  _Float16 v8h;
typedef __attribute__((ext_vector_type(8)))  float    v8f;

#define NNODE 6000
#define MNB   12
#define EF    64      // STEPS == feature width
#define CIN   192     // 3*EF concat width
#define HF    128
#define NROW  144     // MNB*MNB

__device__ __forceinline__ float sp_f(float x) { return x > 15.f ? x : log1pf(__expf(x)); }
__device__ __forceinline__ float lrelu(float x) { return x > 0.f ? x : 0.01f * x; }

__device__ __forceinline__ float wsum(float v) {
#pragma unroll
  for (int m = 16; m; m >>= 1) v += __shfl_xor(v, m, 32);
  return v;
}
__device__ __forceinline__ float wmaxf(float v) {
#pragma unroll
  for (int m = 16; m; m >>= 1) v = fmaxf(v, __shfl_xor(v, m, 32));
  return v;
}

// gfx1250 async global->LDS copy (ASYNCcnt-tracked, bypasses VGPRs).
__device__ __forceinline__ void async_b128(void* lds, const void* g) {
  uint32_t l = (uint32_t)(uintptr_t)lds;   // LDS aperture: addr[31:0] = LDS offset
  uint64_t a = (uint64_t)(uintptr_t)g;
  asm volatile("global_load_async_to_lds_b128 %0, %1, off" :: "v"(l), "v"(a) : "memory");
}
__device__ __forceinline__ void async_wait0() {
  asm volatile("s_wait_asynccnt 0x0" ::: "memory");
}

// A operand (16x32 f16) from row-major LDS [rows][ldk].
// Lane layout: row = lane&15, halves e = K = (lane>>4)*16 + e (contiguous, matches B).
__device__ __forceinline__ v16h load_a(const _Float16* base, int ldk, int rowTile, int k0, int lane) {
  int row = rowTile * 16 + (lane & 15);
  int kh  = (lane >> 4) * 16;
  const _Float16* p = base + row * ldk + k0 + kh;
  v8h lo = *(const v8h*)p;
  v8h hi = *(const v8h*)(p + 8);
  v16h r;
#pragma unroll
  for (int i = 0; i < 8; i++) { r[i] = lo[i]; r[i + 8] = hi[i]; }
  return r;
}

// B operand (32x16 f16) from transposed weights Wt[out][k] in LDS.
// Lane layout: col = lane&15, halves e = K = (lane>>4)*16 + e -> 32 contiguous B.
__device__ __forceinline__ v16h load_b(const _Float16* wt, int ldk, int colTile, int k0, int lane) {
  int o  = colTile * 16 + (lane & 15);
  int kb = k0 + (lane >> 4) * 16;
  return *(const v16h*)(wt + o * ldk + kb);
}

__device__ __forceinline__ v8f wmma_f16(v16h a, v16h b, v8f c) {
  return __builtin_amdgcn_wmma_f32_16x16x32_f16(false, a, false, b, (short)0, c, false, false);
}

// ---------------------------------------------------------------------------
// Weight prep: f32 [din,dout] -> f16 transposed [dout][din]
// ---------------------------------------------------------------------------
__global__ void prep_w_kernel(const float* __restrict__ w, _Float16* __restrict__ out,
                              int din, int dout) {
  int t = blockIdx.x * blockDim.x + threadIdx.x;
  if (t < din * dout) { int k = t / dout, o = t % dout; out[o * din + k] = (_Float16)w[t]; }
}

// ---------------------------------------------------------------------------
// Gaussian basis features -> f16
// ---------------------------------------------------------------------------
__global__ void gbf_kernel(const float* __restrict__ bond, const float* __restrict__ angf,
                           const float* __restrict__ fb, const float* __restrict__ fa,
                           _Float16* __restrict__ edgeOut, _Float16* __restrict__ angOut) {
  const long NE = (long)NNODE * MNB * EF;
  const long NA = (long)NNODE * NROW * EF;
  long stride = (long)gridDim.x * blockDim.x;
  for (long t = (long)blockIdx.x * blockDim.x + threadIdx.x; t < NA; t += stride) {
    int f = (int)(t & 63); long r = t >> 6;
    float d = angf[r] - fa[f];
    angOut[t] = (_Float16)__expf(-d * d * 1024.0f);    // 1/GAMMA_A^2
    if (t < NE) {
      float d2 = bond[r] - fb[f];
      edgeOut[t] = (_Float16)__expf(-d2 * d2 * 64.0f); // 1/GAMMA_B^2
    }
  }
}

// ---------------------------------------------------------------------------
// ConvEdge: one node per block. cat[144x192] @ Wlin[192x64] + att, softmax,
// LN+softplus, sum over j, residual, LN+softplus -> edge_out (f16).
// ---------------------------------------------------------------------------
__global__ __launch_bounds__(256) void conv_edge_kernel(
    const _Float16* __restrict__ edge_in, _Float16* __restrict__ edge_out,
    const _Float16* __restrict__ angle, const int* __restrict__ nbr_idx,
    const _Float16* __restrict__ linWt, const _Float16* __restrict__ attWt,
    const float* __restrict__ lin_b, const float* __restrict__ att_b,
    const float* __restrict__ g1, const float* __restrict__ b1,
    const float* __restrict__ g2, const float* __restrict__ b2) {
  __shared__ __align__(64) _Float16 sCat[NROW * CIN];
  __shared__ __align__(64) _Float16 sBlin[EF * CIN];
  __shared__ __align__(64) _Float16 sBatt[16 * CIN];
  __shared__ __align__(64) float    sH[NROW * EF];
  __shared__ float sAtt[NROW];
  __shared__ float sLb[EF];
  __shared__ float sAb;
  __shared__ int   sNbr[MNB];

  const int n = blockIdx.x, tid = threadIdx.x;
  const int wave = tid >> 5, lane = tid & 31;

  // ---- phase A: LDS tables + async weight staging -------------------------
  if (tid < MNB) sNbr[tid] = nbr_idx[n * MNB + tid];
  if (tid < EF)  sLb[tid] = lin_b[tid];
  if (tid == 0)  sAb = att_b[0];
  for (int t = tid; t < (EF * CIN * 2) / 16; t += 256)          // 1536 chunks
    async_b128((char*)sBlin + t * 16, (const char*)linWt + t * 16);
  if (tid < (CIN * 2) / 16)                                     // 24 chunks
    async_b128((char*)sBatt + tid * 16, (const char*)attWt + tid * 16);
  for (int t = tid; t < 16 * CIN / 2; t += 256)                 // zero pad rows 1..15
    if (t >= CIN / 2) ((uint32_t*)sBatt)[t] = 0u;
  __syncthreads();  // sNbr visible (async copies keep flowing)

  // ---- phase B: async cat staging (24 x 16B chunks per row) ---------------
  for (int t = tid; t < NROW * 24; t += 256) {
    int r = t / 24, c = t - (t / 24) * 24;
    int i = r / MNB, j = r - (r / MNB) * MNB;
    const char* src;
    if (c < 8)       src = (const char*)(edge_in + ((size_t)n * MNB + i) * EF) + c * 16;
    else if (c < 16) src = (const char*)(edge_in + ((size_t)sNbr[i] * MNB + j) * EF) + (c - 8) * 16;
    else             src = (const char*)(angle + ((size_t)n * NROW + r) * EF) + (c - 16) * 16;
    async_b128((char*)sCat + r * (CIN * 2) + c * 16, src);
  }
  async_wait0();
  __syncthreads();

  // ---- WMMA: 36 lin tiles + 9 att tiles -----------------------------------
  const int col = lane & 15, rhi = (lane >> 4) * 8;
  for (int idx = wave; idx < 45; idx += 8) {
    int rt, ct; const _Float16* bb;
    if (idx < 36) { rt = idx >> 2; ct = idx & 3; bb = sBlin; }
    else          { rt = idx - 36; ct = 0;       bb = sBatt; }
    v8f c = {};
#pragma unroll
    for (int ks = 0; ks < 6; ks++) {
      v16h a = load_a(sCat, CIN, rt, ks * 32, lane);
      v16h b = load_b(bb,   CIN, ct, ks * 32, lane);
      c = wmma_f16(a, b, c);
    }
    if (idx < 36) {
      float bias = sLb[ct * 16 + col];
#pragma unroll
      for (int v = 0; v < 8; v++) sH[(rt * 16 + rhi + v) * EF + ct * 16 + col] = c[v] + bias;
    } else if (col == 0) {
      float bias = sAb;
#pragma unroll
      for (int v = 0; v < 8; v++) sAtt[rt * 16 + rhi + v] = c[v] + bias;
    }
  }
  __syncthreads();

  // ---- epilogue: softmax over j, LN+softplus, sum, residual, LN+softplus --
  const float g1l0 = g1[lane], g1l1 = g1[lane + 32], b1l0 = b1[lane], b1l1 = b1[lane + 32];
  const float g2l0 = g2[lane], g2l1 = g2[lane + 32], b2l0 = b2[lane], b2l1 = b2[lane + 32];
  for (int i = wave; i < MNB; i += 8) {
    float av = (lane < MNB) ? lrelu(sAtt[i * MNB + lane]) : -1e30f;
    float mx = wmaxf(av);
    float ex = (lane < MNB) ? __expf(av - mx) : 0.f;
    float se = wsum(ex);
    float alph = ex / se;
    float acc0 = 0.f, acc1 = 0.f;
    for (int j = 0; j < MNB; j++) {
      float aj = __shfl(alph, j, 32);
      int r = i * MNB + j;
      float x0 = aj * sH[r * EF + lane];
      float x1 = aj * sH[r * EF + lane + 32];
      float mu = wsum(x0 + x1) * (1.f / EF);
      float d0 = x0 - mu, d1 = x1 - mu;
      float var = wsum(d0 * d0 + d1 * d1) * (1.f / EF);
      float inv = rsqrtf(var + 1e-5f);
      acc0 += sp_f(d0 * inv * g1l0 + b1l0);
      acc1 += sp_f(d1 * inv * g1l1 + b1l1);
    }
    // residual: this node's edge row still lives in sCat columns [0,64)
    float e0 = (float)sCat[(i * MNB) * CIN + lane] + acc0;
    float e1 = (float)sCat[(i * MNB) * CIN + lane + 32] + acc1;
    float mu = wsum(e0 + e1) * (1.f / EF);
    float d0 = e0 - mu, d1 = e1 - mu;
    float var = wsum(d0 * d0 + d1 * d1) * (1.f / EF);
    float inv = rsqrtf(var + 1e-5f);
    edge_out[((size_t)n * MNB + i) * EF + lane]      = (_Float16)sp_f(d0 * inv * g2l0 + b2l0);
    edge_out[((size_t)n * MNB + i) * EF + lane + 32] = (_Float16)sp_f(d1 * inv * g2l1 + b2l1);
  }
}

// ---------------------------------------------------------------------------
// ConvAngle: in-place angle update (angle is node-local).
// ---------------------------------------------------------------------------
__global__ __launch_bounds__(256) void conv_angle_kernel(
    _Float16* __restrict__ angle, const _Float16* __restrict__ edge_in,
    const int* __restrict__ nbr_idx,
    const _Float16* __restrict__ linWt, const _Float16* __restrict__ attWt,
    const float* __restrict__ lin_b, const float* __restrict__ att_b,
    const float* __restrict__ g2, const float* __restrict__ b2) {
  __shared__ __align__(64) _Float16 sCat[NROW * CIN];
  __shared__ __align__(64) _Float16 sBlin[EF * CIN];
  __shared__ __align__(64) _Float16 sBatt[16 * CIN];
  __shared__ __align__(64) float    sH[NROW * EF];
  __shared__ float sAtt[NROW];
  __shared__ float sLb[EF];
  __shared__ float sAb;
  __shared__ int   sNbr[MNB];

  const int n = blockIdx.x, tid = threadIdx.x;
  const int wave = tid >> 5, lane = tid & 31;

  if (tid < MNB) sNbr[tid] = nbr_idx[n * MNB + tid];
  if (tid < EF)  sLb[tid] = lin_b[tid];
  if (tid == 0)  sAb = att_b[0];
  for (int t = tid; t < (EF * CIN * 2) / 16; t += 256)
    async_b128((char*)sBlin + t * 16, (const char*)linWt + t * 16);
  if (tid < (CIN * 2) / 16)
    async_b128((char*)sBatt + tid * 16, (const char*)attWt + tid * 16);
  for (int t = tid; t < 16 * CIN / 2; t += 256)
    if (t >= CIN / 2) ((uint32_t*)sBatt)[t] = 0u;
  __syncthreads();

  for (int t = tid; t < NROW * 24; t += 256) {
    int r = t / 24, c = t - (t / 24) * 24;
    int i = r / MNB, j = r - (r / MNB) * MNB;
    const char* src;
    if (c < 8)       src = (const char*)(edge_in + ((size_t)n * MNB + i) * EF) + c * 16;
    else if (c < 16) src = (const char*)(edge_in + ((size_t)sNbr[i] * MNB + j) * EF) + (c - 8) * 16;
    else             src = (const char*)(angle + ((size_t)n * NROW + r) * EF) + (c - 16) * 16;
    async_b128((char*)sCat + r * (CIN * 2) + c * 16, src);
  }
  async_wait0();
  __syncthreads();

  const int col = lane & 15, rhi = (lane >> 4) * 8;
  for (int idx = wave; idx < 45; idx += 8) {
    int rt, ct; const _Float16* bb;
    if (idx < 36) { rt = idx >> 2; ct = idx & 3; bb = sBlin; }
    else          { rt = idx - 36; ct = 0;       bb = sBatt; }
    v8f c = {};
#pragma unroll
    for (int ks = 0; ks < 6; ks++) {
      v16h a = load_a(sCat, CIN, rt, ks * 32, lane);
      v16h b = load_b(bb,   CIN, ct, ks * 32, lane);
      c = wmma_f16(a, b, c);
    }
    if (idx < 36) {
      float bias = sLb[ct * 16 + col];
#pragma unroll
      for (int v = 0; v < 8; v++) sH[(rt * 16 + rhi + v) * EF + ct * 16 + col] = c[v] + bias;
    } else if (col == 0) {
      float bias = sAb;
#pragma unroll
      for (int v = 0; v < 8; v++) sAtt[rt * 16 + rhi + v] = c[v] + bias;
    }
  }
  __syncthreads();

  const float g2l0 = g2[lane], g2l1 = g2[lane + 32], b2l0 = b2[lane], b2l1 = b2[lane + 32];
  for (int r = wave; r < NROW; r += 8) {
    float al = lrelu(sAtt[r]);
    // original angle value still lives in sCat columns [128,192)
    float x0 = (float)sCat[r * CIN + 2 * EF + lane]      + al * sH[r * EF + lane];
    float x1 = (float)sCat[r * CIN + 2 * EF + lane + 32] + al * sH[r * EF + lane + 32];
    float mu = wsum(x0 + x1) * (1.f / EF);
    float d0 = x0 - mu, d1 = x1 - mu;
    float var = wsum(d0 * d0 + d1 * d1) * (1.f / EF);
    float inv = rsqrtf(var + 1e-5f);
    angle[((size_t)n * NROW + r) * EF + lane]      = (_Float16)sp_f(d0 * inv * g2l0 + b2l0);
    angle[((size_t)n * NROW + r) * EF + lane + 32] = (_Float16)sp_f(d1 * inv * g2l1 + b2l1);
  }
}

// ---------------------------------------------------------------------------
// Final: expand (64->128 WMMA), softplus/sum pools, LN(256), out (256->2).
// ---------------------------------------------------------------------------
__global__ __launch_bounds__(256) void final_kernel(
    const _Float16* __restrict__ edge, const _Float16* __restrict__ angle,
    const _Float16* __restrict__ WeT, const _Float16* __restrict__ WaT,
    const float* __restrict__ be, const float* __restrict__ ba,
    const float* __restrict__ gbn, const float* __restrict__ bbn,
    const float* __restrict__ outw, const float* __restrict__ outb,
    float* __restrict__ out) {
  __shared__ __align__(64) _Float16 sAng[NROW * EF];
  __shared__ __align__(64) _Float16 sEdge[16 * EF];     // rows 12..15 zero-padded
  __shared__ __align__(64) _Float16 sWa[HF * EF];
  __shared__ __align__(64) _Float16 sWe[HF * EF];
  __shared__ __align__(64) float sP[NROW * HF];
  __shared__ __align__(64) float sE[16 * HF];
  __shared__ float sQ[MNB * HF];
  __shared__ float sCrys[2 * HF];
  __shared__ float sBa[HF], sBe[HF];
  __shared__ float sRed[8];

  const int n = blockIdx.x, tid = threadIdx.x;
  const int wave = tid >> 5, lane = tid & 31;

  for (int t = tid; t < (NROW * EF * 2) / 16; t += 256)          // 1152 chunks
    async_b128((char*)sAng + t * 16, (const char*)(angle + (size_t)n * NROW * EF) + t * 16);
  if (tid < (MNB * EF * 2) / 16)                                 // 96 chunks
    async_b128((char*)sEdge + tid * 16, (const char*)(edge + (size_t)n * MNB * EF) + tid * 16);
  for (int t = tid; t < 16 * EF / 2; t += 256)                   // zero pad rows 12..15
    if (t >= MNB * EF / 2) ((uint32_t*)sEdge)[t] = 0u;
  for (int t = tid; t < (HF * EF * 2) / 16; t += 256) {          // 1024 chunks each
    async_b128((char*)sWa + t * 16, (const char*)WaT + t * 16);
    async_b128((char*)sWe + t * 16, (const char*)WeT + t * 16);
  }
  if (tid < HF) { sBa[tid] = ba[tid]; sBe[tid] = be[tid]; }
  async_wait0();
  __syncthreads();

  const int col = lane & 15, rhi = (lane >> 4) * 8;
  for (int idx = wave; idx < 80; idx += 8) {  // 72 angle tiles + 8 edge tiles
    v8f c = {};
    if (idx < 72) {
      int rt = idx >> 3, ct = idx & 7;
#pragma unroll
      for (int ks = 0; ks < 2; ks++) {
        v16h a = load_a(sAng, EF, rt, ks * 32, lane);
        v16h b = load_b(sWa, EF, ct, ks * 32, lane);
        c = wmma_f16(a, b, c);
      }
      float bias = sBa[ct * 16 + col];
#pragma unroll
      for (int v = 0; v < 8; v++) sP[(rt * 16 + rhi + v) * HF + ct * 16 + col] = c[v] + bias;
    } else {
      int ct = idx - 72;
#pragma unroll
      for (int ks = 0; ks < 2; ks++) {
        v16h a = load_a(sEdge, EF, 0, ks * 32, lane);
        v16h b = load_b(sWe, EF, ct, ks * 32, lane);
        c = wmma_f16(a, b, c);
      }
      float bias = sBe[ct * 16 + col];
#pragma unroll
      for (int v = 0; v < 8; v++) sE[(rhi + v) * HF + ct * 16 + col] = c[v] + bias;
    }
  }
  __syncthreads();

  for (int t = tid; t < MNB * HF; t += 256) {
    int i = t >> 7, c = t & 127;
    float s = 0.f;
#pragma unroll
    for (int j = 0; j < MNB; j++) s += sp_f(sP[(i * MNB + j) * HF + c]);
    sQ[t] = s;
  }
  __syncthreads();
  {
    int t = tid;  // exactly 256 threads -> one crys element each
    float v = 0.f;
    if (t < HF) {
#pragma unroll
      for (int i = 0; i < MNB; i++) v += sp_f(sE[i * HF + t]);
    } else {
      int c = t - HF;
#pragma unroll
      for (int i = 0; i < MNB; i++) v += sp_f(sQ[i * HF + c]);
    }
    sCrys[t] = v;
  }
  __syncthreads();

  float x = sCrys[tid];
  float s = wsum(x);
  if (lane == 0) sRed[wave] = s;
  __syncthreads();
  float tot = 0.f;
#pragma unroll
  for (int w = 0; w < 8; w++) tot += sRed[w];
  float mu = tot * (1.f / 256.f);
  float d = x - mu;
  float q = wsum(d * d);
  __syncthreads();
  if (lane == 0) sRed[wave] = q;
  __syncthreads();
  float vq = 0.f;
#pragma unroll
  for (int w = 0; w < 8; w++) vq += sRed[w];
  float inv = rsqrtf(vq * (1.f / 256.f) + 1e-5f);
  sCrys[tid] = sp_f(d * inv * gbn[tid] + bbn[tid]);
  __syncthreads();

  if (tid < 2) {
    float o = outb[tid];
    for (int k = 0; k < 2 * HF; k++) o += sCrys[k] * outw[k * 2 + tid];
    out[(size_t)n * 2 + tid] = o;
  }
}

// ---------------------------------------------------------------------------
// Host-side launcher
// ---------------------------------------------------------------------------
extern "C" void kernel_launch(void* const* d_in, const int* in_sizes, int n_in,
                              void* d_out, int out_size, void* d_ws, size_t ws_size,
                              hipStream_t stream) {
  (void)in_sizes; (void)n_in; (void)out_size; (void)ws_size;
  const float* bond = (const float*)d_in[0];
  const float* angf = (const float*)d_in[1];
  const int*   nbr  = (const int*)d_in[3];
  const float* fb   = (const float*)d_in[5];
  const float* fa   = (const float*)d_in[6];
  // d_in layout (insertion order of setup_inputs / params):
  //   7 + 8*l : edge conv l -> lin.w, lin.b, att.w, att.b, g1, b1, g2, b2
  //   31 + 6*l: ang  conv l -> lin.w, lin.b, att.w, att.b, g2, b2
  //   43/44 expand_edge w/b, 45/46 expand_angle w/b, 47 g_bn, 48 b_bn, 49/50 out w/b

  char* ws = (char*)d_ws;
  size_t off = 0;
  auto take = [&](size_t bytes) -> char* {
    char* p = ws + off;
    off += (bytes + 255) & ~(size_t)255;
    return p;
  };
  _Float16* wEL[3]; _Float16* wEA[3];
  for (int l = 0; l < 3; l++) { wEL[l] = (_Float16*)take(EF * CIN * 2); wEA[l] = (_Float16*)take(CIN * 2); }
  _Float16* wAL[2]; _Float16* wAA[2];
  for (int l = 0; l < 2; l++) { wAL[l] = (_Float16*)take(EF * CIN * 2); wAA[l] = (_Float16*)take(CIN * 2); }
  _Float16* wExpE = (_Float16*)take((size_t)HF * EF * 2);
  _Float16* wExpA = (_Float16*)take((size_t)HF * EF * 2);
  _Float16* edgeA = (_Float16*)take((size_t)NNODE * MNB * EF * 2);
  _Float16* edgeB = (_Float16*)take((size_t)NNODE * MNB * EF * 2);
  _Float16* angB  = (_Float16*)take((size_t)NNODE * NROW * EF * 2);

  auto prep = [&](const float* w, _Float16* o, int din, int dout) {
    int ne = din * dout;
    prep_w_kernel<<<(ne + 255) / 256, 256, 0, stream>>>(w, o, din, dout);
  };
  for (int l = 0; l < 3; l++) {
    prep((const float*)d_in[7 + 8 * l + 0], wEL[l], CIN, EF);
    prep((const float*)d_in[7 + 8 * l + 2], wEA[l], CIN, 1);
  }
  for (int l = 0; l < 2; l++) {
    prep((const float*)d_in[31 + 6 * l + 0], wAL[l], CIN, EF);
    prep((const float*)d_in[31 + 6 * l + 2], wAA[l], CIN, 1);
  }
  prep((const float*)d_in[43], wExpE, EF, HF);
  prep((const float*)d_in[45], wExpA, EF, HF);

  long NA = (long)NNODE * NROW * EF;
  gbf_kernel<<<(int)((NA + 255) / 256), 256, 0, stream>>>(bond, angf, fb, fa, edgeA, angB);

  auto convE = [&](const _Float16* ein, _Float16* eout, int l) {
    conv_edge_kernel<<<NNODE, 256, 0, stream>>>(
        ein, eout, angB, nbr, wEL[l], wEA[l],
        (const float*)d_in[7 + 8 * l + 1], (const float*)d_in[7 + 8 * l + 3],
        (const float*)d_in[7 + 8 * l + 4], (const float*)d_in[7 + 8 * l + 5],
        (const float*)d_in[7 + 8 * l + 6], (const float*)d_in[7 + 8 * l + 7]);
  };
  auto convA = [&](const _Float16* ein, int l) {
    conv_angle_kernel<<<NNODE, 256, 0, stream>>>(
        angB, ein, nbr, wAL[l], wAA[l],
        (const float*)d_in[31 + 6 * l + 1], (const float*)d_in[31 + 6 * l + 3],
        (const float*)d_in[31 + 6 * l + 4], (const float*)d_in[31 + 6 * l + 5]);
  };

  convE(edgeA, edgeB, 0);
  convA(edgeB, 0);
  convE(edgeB, edgeA, 1);
  convA(edgeA, 1);
  convE(edgeA, edgeB, 2);

  final_kernel<<<NNODE, 256, 0, stream>>>(
      edgeB, angB, wExpE, wExpA,
      (const float*)d_in[44], (const float*)d_in[46],
      (const float*)d_in[47], (const float*)d_in[48],
      (const float*)d_in[49], (const float*)d_in[50],
      (float*)d_out);
}
